// GTLayer_76647986364559
// MI455X (gfx1250) — compile-verified, matrix operand order checked by reading, not run on hardware
//
#include <hip/hip_runtime.h>
#include <hip/hip_bf16.h>
#include <cstdint>
#include <cstddef>

#define B_  4
#define N_  512
#define D_  256
#define H_  8
#define HD_ 32

typedef __bf16 bf16;
typedef __attribute__((ext_vector_type(16))) __bf16 v16bf;
typedef __attribute__((ext_vector_type(8)))  float  v8f;

static __device__ __forceinline__ v8f v8f_zero() {
  v8f v;
#pragma unroll
  for (int i = 0; i < 8; ++i) v[i] = 0.0f;
  return v;
}

// Load one 16x32 bf16 WMMA operand tile. Both A and B operands are
// K-contiguous row-major here (weights are stored (out,in)), so the same
// lane pattern serves both: lane l<16 holds row (l) K={0..7,16..23} of the
// 32-wide K slab, lane l>=16 holds row (l-15... l&15) K={8..15,24..31}.
static __device__ __forceinline__ v16bf load_tile_b16(const bf16* base, int ld, int row0, int k0) {
  const int lane = threadIdx.x & 31;
  const bf16* p = base + (size_t)(row0 + (lane & 15)) * ld + (size_t)(k0 + ((lane >> 4) << 3));
  union { v16bf v; uint4 q[2]; } u;
  u.q[0] = *(const uint4*)(p);
  u.q[1] = *(const uint4*)(p + 16);
  return u.v;
}

// Same tile pattern, but source is f32 (global e tensor or LDS scores);
// convert to bf16 in-register. Keeps HBM traffic fp32 (inputs are fp32)
// while letting the matmul ride the bf16 WMMA pipe.
static __device__ __forceinline__ v16bf load_tile_f32cvt(const float* base, int ld, int row0, int k0) {
  const int lane = threadIdx.x & 31;
  const float* p = base + (size_t)(row0 + (lane & 15)) * ld + (size_t)(k0 + ((lane >> 4) << 3));
  float4 a0 = *(const float4*)(p);
  float4 a1 = *(const float4*)(p + 4);
  float4 a2 = *(const float4*)(p + 16);
  float4 a3 = *(const float4*)(p + 20);
  v16bf v;
  v[0]=(bf16)a0.x;  v[1]=(bf16)a0.y;  v[2]=(bf16)a0.z;  v[3]=(bf16)a0.w;
  v[4]=(bf16)a1.x;  v[5]=(bf16)a1.y;  v[6]=(bf16)a1.z;  v[7]=(bf16)a1.w;
  v[8]=(bf16)a2.x;  v[9]=(bf16)a2.y;  v[10]=(bf16)a2.z; v[11]=(bf16)a2.w;
  v[12]=(bf16)a3.x; v[13]=(bf16)a3.y; v[14]=(bf16)a3.z; v[15]=(bf16)a3.w;
  return v;
}

// ---------------------------------------------------------------- cvt f32->bf16
__global__ void cvt_f32_bf16_kernel(const float* __restrict__ in, bf16* __restrict__ out, int n) {
  int i = blockIdx.x * blockDim.x + threadIdx.x;
  if (i < n) out[i] = (bf16)in[i];
}

// ---------------------------------------------------------------- RMSNorm -> bf16
__global__ void rmsnorm_kernel(const float* __restrict__ in, const float* __restrict__ w,
                               bf16* __restrict__ out, int rows) {
  const int cols = D_;
  int wave = threadIdx.x >> 5, lane = threadIdx.x & 31;
  int row = blockIdx.x * (blockDim.x >> 5) + wave;
  if (row >= rows) return;
  const float* p = in + (size_t)row * cols;
  float vals[8]; float ss = 0.0f;
#pragma unroll
  for (int t = 0; t < 8; ++t) { float v = p[lane + 32 * t]; vals[t] = v; ss += v * v; }
#pragma unroll
  for (int off = 16; off; off >>= 1) ss += __shfl_xor(ss, off, 32);
  float r = rsqrtf(ss / (float)cols + 1e-5f);
#pragma unroll
  for (int t = 0; t < 8; ++t)
    out[(size_t)row * cols + lane + 32 * t] = (bf16)(vals[t] * r * w[lane + 32 * t]);
}

// ------------------------------------------- residual add + RMSNorm (y f32, z bf16)
__global__ void resnorm_kernel(const float* __restrict__ yattn, const float* __restrict__ hres,
                               const float* __restrict__ w, float* __restrict__ yout,
                               bf16* __restrict__ z, int rows) {
  const int cols = D_;
  int wave = threadIdx.x >> 5, lane = threadIdx.x & 31;
  int row = blockIdx.x * (blockDim.x >> 5) + wave;
  if (row >= rows) return;
  size_t base = (size_t)row * cols;
  float vals[8]; float ss = 0.0f;
#pragma unroll
  for (int t = 0; t < 8; ++t) {
    float v = yattn[base + lane + 32 * t] + hres[base + lane + 32 * t];
    yout[base + lane + 32 * t] = v;
    vals[t] = v; ss += v * v;
  }
#pragma unroll
  for (int off = 16; off; off >>= 1) ss += __shfl_xor(ss, off, 32);
  float r = rsqrtf(ss / (float)cols + 1e-5f);
#pragma unroll
  for (int t = 0; t < 8; ++t)
    z[base + lane + 32 * t] = (bf16)(vals[t] * r * w[lane + 32 * t]);
}

// ------------------------------------------------------- generic bf16 WMMA GEMM
// C(MxN) = A(MxK) * Bw(NxK)^T.  4 waves/block, each wave: 16 rows x 64 cols.
// mode 0: store f32; mode 1: exact GELU -> bf16; mode 2: +resid -> f32.
__global__ __launch_bounds__(128) void gemm_bf16_kernel(
    const bf16* __restrict__ A, const bf16* __restrict__ Bw,
    int Ntot, int K, int mode,
    const float* __restrict__ resid, float* __restrict__ outf, bf16* __restrict__ outb) {
  int wave = threadIdx.x >> 5, lane = threadIdx.x & 31;
  int row0 = (blockIdx.y * 4 + wave) * 16;
  int col0 = blockIdx.x * 64;
  v8f acc0 = v8f_zero(), acc1 = v8f_zero(), acc2 = v8f_zero(), acc3 = v8f_zero();
  for (int kk = 0; kk < K; kk += 32) {
    v16bf a  = load_tile_b16(A, K, row0, kk);
    v16bf b0 = load_tile_b16(Bw, K, col0 +  0, kk);
    v16bf b1 = load_tile_b16(Bw, K, col0 + 16, kk);
    v16bf b2 = load_tile_b16(Bw, K, col0 + 32, kk);
    v16bf b3 = load_tile_b16(Bw, K, col0 + 48, kk);
    acc0 = __builtin_amdgcn_wmma_f32_16x16x32_bf16(false, a, false, b0, (short)0, acc0, false, false);
    acc1 = __builtin_amdgcn_wmma_f32_16x16x32_bf16(false, a, false, b1, (short)0, acc1, false, false);
    acc2 = __builtin_amdgcn_wmma_f32_16x16x32_bf16(false, a, false, b2, (short)0, acc2, false, false);
    acc3 = __builtin_amdgcn_wmma_f32_16x16x32_bf16(false, a, false, b3, (short)0, acc3, false, false);
  }
  int mrow = (lane >> 4) << 3, ncol = lane & 15;
  v8f accs[4] = {acc0, acc1, acc2, acc3};
#pragma unroll
  for (int ct = 0; ct < 4; ++ct) {
    int col = col0 + ct * 16 + ncol;
#pragma unroll
    for (int r = 0; r < 8; ++r) {
      size_t idx = (size_t)(row0 + mrow + r) * Ntot + col;
      float v = accs[ct][r];
      if (mode == 0) {
        outf[idx] = v;
      } else if (mode == 1) {
        outb[idx] = (bf16)(0.5f * v * (1.0f + erff(v * 0.70710678118654752f)));
      } else {
        outf[idx] = v + resid[idx];
      }
    }
  }
}

// ----------------------------------------------- e-projection (the HBM-bound op)
// e12(M x 16) = e(M x 256) * We(16 x 256)^T,  M = B*N*N.  Each wave: 16 rows.
__global__ __launch_bounds__(128) void eproj_kernel(
    const float* __restrict__ e, const bf16* __restrict__ We, float* __restrict__ e12) {
  int wave = threadIdx.x >> 5, lane = threadIdx.x & 31;
  int row0 = (blockIdx.x * 4 + wave) * 16;
  v8f acc = v8f_zero();
#pragma unroll
  for (int kk = 0; kk < D_; kk += 32) {
    v16bf a = load_tile_f32cvt(e, D_, row0, kk);   // fp32 HBM reads, cvt in-reg
    v16bf b = load_tile_b16(We, D_, 0, kk);        // 8 KB weight: L2/WGP$ hot
    acc = __builtin_amdgcn_wmma_f32_16x16x32_bf16(false, a, false, b, (short)0, acc, false, false);
  }
  int mrow = (lane >> 4) << 3, ncol = lane & 15;
#pragma unroll
  for (int r = 0; r < 8; ++r)
    e12[(size_t)(row0 + mrow + r) * 16 + ncol] = acc[r];
}

// ------------------------------------------- qkv f32 -> q,k bf16 (B,H,N,hd), vT (B,H,hd,N)
__global__ void qkv_reshape_kernel(const float* __restrict__ qkv,
                                   bf16* __restrict__ q, bf16* __restrict__ k, bf16* __restrict__ vT) {
  int idx = blockIdx.x * blockDim.x + threadIdx.x;
  const int total = B_ * N_ * 3 * D_;
  if (idx >= total) return;
  int c = idx % (3 * D_);
  int n = (idx / (3 * D_)) % N_;
  int b = idx / (3 * D_ * N_);
  int which = c / D_;
  int d = c % D_;
  int h = d / HD_, dd = d % HD_;
  float val = qkv[idx];
  size_t bh = (size_t)(b * H_ + h);
  if (which == 0)      q[(bh * N_ + n) * HD_ + dd] = (bf16)val;
  else if (which == 1) k[(bh * N_ + n) * HD_ + dd] = (bf16)val;
  else                 vT[(bh * HD_ + dd) * N_ + n] = (bf16)val;
}

// -------------------------------------------------------------------- attention
// One wave per (b,h, 16-row i-tile).  S = q k^T (WMMA) + e1 -> LDS; row softmax;
// e2 gate folded into P, 1/sum folded into final scale; y = P V via WMMA on vT.
#define SCLD 516   // 16B-aligned pad: 2064B row stride -> 4-bank lane stride
__global__ __launch_bounds__(32) void attn_kernel(
    const bf16* __restrict__ q, const bf16* __restrict__ k, const bf16* __restrict__ vT,
    const float* __restrict__ e12, float* __restrict__ yattn) {
  __shared__ float sc[16][SCLD];
  __shared__ float sinv[16];
  int bh = blockIdx.y;
  int b = bh >> 3, h = bh & 7;
  int i0 = blockIdx.x * 16;
  int lane = threadIdx.x;
  int mrow = (lane >> 4) << 3, ncol = lane & 15;
  const bf16* qb = q  + (size_t)bh * N_ * HD_;
  const bf16* kb = k  + (size_t)bh * N_ * HD_;
  const bf16* vb = vT + (size_t)bh * HD_ * N_;
  const float scale = 0.17677669529663688f;  // 1/sqrt(32)

  v16bf qa = load_tile_b16(qb, HD_, i0, 0);  // K = hd = 32: single slab
  for (int jt = 0; jt < N_ / 16; ++jt) {
    int j0 = jt * 16;
    v16bf kt = load_tile_b16(kb, HD_, j0, 0);
    v8f s = v8f_zero();
    s = __builtin_amdgcn_wmma_f32_16x16x32_bf16(false, qa, false, kt, (short)0, s, false, false);
#pragma unroll
    for (int r = 0; r < 8; ++r) {
      int i = i0 + mrow + r, j = j0 + ncol;
      float e1 = e12[(((size_t)b * N_ + i) * N_ + j) * (2 * H_) + h];
      sc[mrow + r][j] = s[r] * scale + e1;
    }
  }
  __syncthreads();

  // two lanes per row: lane = 2*row + half, each scans 256 entries
  int row = lane >> 1, half = lane & 1;
  int gi = i0 + row;
  float mx = -3.0e38f;
  for (int j = half * 256; j < half * 256 + 256; ++j) mx = fmaxf(mx, sc[row][j]);
  mx = fmaxf(mx, __shfl_xor(mx, 1, 32));
  float sum = 0.0f;
  for (int j = half * 256; j < half * 256 + 256; ++j) {
    float ex = __expf(sc[row][j] - mx);
    sum += ex;                                    // denominator excludes e2
    float e2 = e12[(((size_t)b * N_ + gi) * N_ + j) * (2 * H_) + H_ + h];
    sc[row][j] = ex * e2;                         // gate folded into P
  }
  sum += __shfl_xor(sum, 1, 32);
  if (half == 0) sinv[row] = 1.0f / sum;
  __syncthreads();

#pragma unroll
  for (int ct = 0; ct < 2; ++ct) {                // two 16-wide hd tiles
    v8f acc = v8f_zero();
    for (int jt = 0; jt < N_ / 32; ++jt) {
      v16bf pa = load_tile_f32cvt(&sc[0][0], SCLD, 0, jt * 32);
      v16bf vm = load_tile_b16(vb, N_, ct * 16, jt * 32);   // vT rows contiguous in j
      acc = __builtin_amdgcn_wmma_f32_16x16x32_bf16(false, pa, false, vm, (short)0, acc, false, false);
    }
#pragma unroll
    for (int r = 0; r < 8; ++r) {
      int i = i0 + mrow + r;
      float yv = acc[r] * sinv[mrow + r];         // 1/sum applied here
      yattn[((size_t)b * N_ + i) * D_ + h * HD_ + ct * 16 + ncol] = yv;
    }
  }
}

// ================================================================= host launcher
extern "C" void kernel_launch(void* const* d_in, const int* in_sizes, int n_in,
                              void* d_out, int out_size, void* d_ws, size_t ws_size,
                              hipStream_t stream) {
  const float* h      = (const float*)d_in[0];
  const float* e      = (const float*)d_in[1];
  const float* W_h    = (const float*)d_in[2];
  const float* W_e    = (const float*)d_in[3];
  const float* norm1w = (const float*)d_in[4];
  const float* norm2w = (const float*)d_in[5];
  const float* c_fc   = (const float*)d_in[6];
  const float* c_proj = (const float*)d_in[7];
  float* out = (float*)d_out;

  char* wsb = (char*)d_ws;
  size_t off = 0;
  auto alloc = [&](size_t bytes) -> void* {
    void* p = wsb + off;
    off = (off + bytes + 255) & ~(size_t)255;
    return p;
  };
  const int M  = B_ * N_;            // 2048
  const int ME = B_ * N_ * N_;       // 1,048,576
  bf16*  Wh_b  = (bf16*)alloc((size_t)3 * D_ * D_ * 2);
  bf16*  We_b  = (bf16*)alloc((size_t)2 * H_ * D_ * 2);
  bf16*  fc_b  = (bf16*)alloc((size_t)4 * D_ * D_ * 2);
  bf16*  pj_b  = (bf16*)alloc((size_t)D_ * 4 * D_ * 2);
  bf16*  x_b   = (bf16*)alloc((size_t)M * D_ * 2);
  float* qkv_f = (float*)alloc((size_t)M * 3 * D_ * 4);
  bf16*  q_b   = (bf16*)alloc((size_t)M * D_ * 2);
  bf16*  k_b   = (bf16*)alloc((size_t)M * D_ * 2);
  bf16*  vT_b  = (bf16*)alloc((size_t)M * D_ * 2);
  float* e12   = (float*)alloc((size_t)ME * 2 * H_ * 4);   // 67 MB
  float* yat   = (float*)alloc((size_t)M * D_ * 4);
  float* yfull = (float*)alloc((size_t)M * D_ * 4);
  bf16*  z_b   = (bf16*)alloc((size_t)M * D_ * 2);
  bf16*  a_b   = (bf16*)alloc((size_t)M * 4 * D_ * 2);
  (void)ws_size; (void)n_in; (void)in_sizes; (void)out_size;

  // 1) weights -> bf16
  {
    int n;
    n = 3 * D_ * D_;  cvt_f32_bf16_kernel<<<(n + 255) / 256, 256, 0, stream>>>(W_h,   Wh_b, n);
    n = 2 * H_ * D_;  cvt_f32_bf16_kernel<<<(n + 255) / 256, 256, 0, stream>>>(W_e,   We_b, n);
    n = 4 * D_ * D_;  cvt_f32_bf16_kernel<<<(n + 255) / 256, 256, 0, stream>>>(c_fc,  fc_b, n);
    n = 4 * D_ * D_;  cvt_f32_bf16_kernel<<<(n + 255) / 256, 256, 0, stream>>>(c_proj, pj_b, n);
  }
  // 2) x = rmsnorm(h)
  rmsnorm_kernel<<<M / 8, 256, 0, stream>>>(h, norm1w, x_b, M);
  // 3) qkv = x @ W_h^T   (2048 x 768 x 256)
  gemm_bf16_kernel<<<dim3(3 * D_ / 64, M / 64), 128, 0, stream>>>(
      x_b, Wh_b, 3 * D_, D_, 0, nullptr, qkv_f, nullptr);
  // 4) e12 = e @ W_e^T   (1M x 16 x 256)  -- HBM-bound hot loop
  eproj_kernel<<<ME / 64, 128, 0, stream>>>(e, We_b, e12);
  // 5) split heads
  {
    int n = M * 3 * D_;
    qkv_reshape_kernel<<<(n + 255) / 256, 256, 0, stream>>>(qkv_f, q_b, k_b, vT_b);
  }
  // 6) attention
  attn_kernel<<<dim3(N_ / 16, B_ * H_), 32, 0, stream>>>(q_b, k_b, vT_b, e12, yat);
  // 7) y = yattn + h ; z = rmsnorm(y)
  resnorm_kernel<<<M / 8, 256, 0, stream>>>(yat, h, norm2w, yfull, z_b, M);
  // 8) MLP: gelu(z @ c_fc^T) -> bf16 ; @ c_proj^T + y -> out
  gemm_bf16_kernel<<<dim3(4 * D_ / 64, M / 64), 128, 0, stream>>>(
      z_b, fc_b, 4 * D_, D_, 1, nullptr, nullptr, a_b);
  gemm_bf16_kernel<<<dim3(D_ / 64, M / 64), 128, 0, stream>>>(
      a_b, pj_b, D_, 4 * D_, 2, yfull, out, nullptr);
}
